// TrueNorthSmall2_26319559590636
// MI455X (gfx1250) — matrix operand order repaired and checked by reading, not compiled, not run on hardware
//
#include <hip/hip_runtime.h>
#include <hip/hip_bf16.h>

typedef __attribute__((ext_vector_type(16))) _Float16 v16h;
typedef __attribute__((ext_vector_type(8)))  _Float16 v8h;
typedef __attribute__((ext_vector_type(8)))  float    v8f;
typedef __attribute__((ext_vector_type(4)))  unsigned int v4u;
typedef __attribute__((ext_vector_type(8)))  int      v8i;
typedef __attribute__((ext_vector_type(4)))  int      v4i;
typedef _Float16 h16;

// ---------------- sizes / workspace layout (element offsets) ----------------
static constexpr int    BB     = 16;      // batch
static constexpr int    TT     = 32;      // timesteps
// float section
static constexpr size_t N_S1   = 76800;   // B*60*80
static constexpr size_t N_S2   = 1228800; // B*16*60*80
static constexpr size_t N_S3   = 307200;  // B*16*30*40
static constexpr size_t N_S4   = 614400;  // B*32*30*40
static constexpr size_t N_S5   = 153600;  // B*32*15*20
static constexpr size_t OF_S1V = 0;
static constexpr size_t OF_S1I = OF_S1V + N_S1;
static constexpr size_t OF_S2V = OF_S1I + N_S1;
static constexpr size_t OF_S2I = OF_S2V + N_S2;
static constexpr size_t OF_S3V = OF_S2I + N_S2;
static constexpr size_t OF_S3I = OF_S3V + N_S3;
static constexpr size_t OF_S4V = OF_S3I + N_S3;
static constexpr size_t OF_S4I = OF_S4V + N_S4;
static constexpr size_t OF_S5V = OF_S4I + N_S4;
static constexpr size_t OF_S5I = OF_S5V + N_S5;
static constexpr size_t OF_SOV = OF_S5I + N_S5;
static constexpr size_t OF_SOI = OF_SOV + 80;
static constexpr size_t OF_FC1A= OF_SOI + 80;      // [16][512] relu(fc1) output
static constexpr size_t NF_TOTAL = OF_FC1A + 16*512;
// half section (spikes + fragment-swizzled f16 weights)
static constexpr size_t OH_Z1  = 0;                 // [B,60,80]
static constexpr size_t OH_Z2  = OH_Z1 + 76800;     // [B,16,60,80]
static constexpr size_t OH_Z3  = OH_Z2 + 614400;    // [B,16,30,40]
static constexpr size_t OH_Z4  = OH_Z3 + 307200;    // [B,32,30,40]
static constexpr size_t OH_Z5  = OH_Z4 + 614400;    // [B,9600] flattened spikes
static constexpr size_t OH_W1F = OH_Z5 + 153600;    // conv1 A-fragments [32lane][16]
static constexpr size_t OH_W2F = OH_W1F + 512;      // conv2 B-fragments [5ks][2nt][32lane][16]
static constexpr size_t OH_FWF = OH_W2F + 5120;     // fc1 B-fragments [32nt][300ks][32lane][16]

// WMMA fragment index maps (CDNA5 ISA 7.12.2, 16-bit A 16x32 / B 32x16)
__device__ __forceinline__ int a_kmap(int e, int lane) {
  return ((e < 8) ? e : e + 8) + ((lane >> 4) << 3);
}
__device__ __forceinline__ int b_kmap(int e, int lane) {
  return ((e < 8) ? e : e + 8) + ((lane >> 4) << 3);
}

// ---------------- prep kernels ----------------
__global__ void k_zero(float* F) {
  size_t i = (size_t)blockIdx.x * blockDim.x + threadIdx.x;
  size_t stride = (size_t)gridDim.x * blockDim.x;
  for (; i < NF_TOTAL; i += stride) F[i] = 0.0f;
}

// pre-swizzle conv1 + conv2 weights into WMMA fragment order
__global__ void k_prep_w(h16* H, const float* __restrict__ c1w,
                         const float* __restrict__ c2w) {
  int i = blockIdx.x * blockDim.x + threadIdx.x;
  if (i < 512) {                       // conv1 A-fragment: lane l holds row M=l&15
    int l = i >> 4, e = i & 15;
    int K = a_kmap(e, l);              // K over 5x5 (pad 25->32)
    int rowM = l & 15;
    H[OH_W1F + i] = (K < 25) ? (h16)c1w[rowM * 25 + K] : (h16)0.f;
  } else if (i < 5632) {               // conv2 B-fragment [ks][nt][l][e]
    int j  = i - 512;
    int e  = j & 15, l = (j >> 4) & 31, nt = (j >> 9) & 1, ks = j >> 10;
    int K  = ks * 32 + b_kmap(e, l);   // K = ci*9+kh*3+kw (pad 144->160)
    int co = nt * 16 + (l & 15);
    H[OH_W2F + j] = (K < 144) ? (h16)c2w[co * 144 + K] : (h16)0.f;
  }
}

// pre-swizzle fc1 weights into per-lane contiguous B fragments [nt][ks][l][e]
__global__ void k_prep_fwt(h16* H, const float* __restrict__ fc1w) {
  size_t i = (size_t)blockIdx.x * blockDim.x + threadIdx.x;
  if (i >= (size_t)32 * 300 * 32 * 16) return;
  int e  = (int)(i & 15);
  int l  = (int)((i >> 4) & 31);
  int r  = (int)(i >> 9);
  int ks = r % 300, nt = r / 300;
  int K  = ks * 32 + b_kmap(e, l);
  int n  = nt * 16 + (l & 15);
  H[OH_FWF + i] = (n < 500) ? (h16)fc1w[(size_t)n * 9600 + K] : (h16)0.f;
}

// ---------------- per-timestep kernels ----------------
// 4x4 maxpool on raw input + LIF1 -> spike z1 (f16)
__global__ void k_pool4_lif1(const float* __restrict__ x, h16* H, float* F, int t) {
  int idx = blockIdx.x * blockDim.x + threadIdx.x;
  if (idx >= 76800) return;
  int b = idx / 4800, r = idx % 4800, y = r / 80, xx = r % 80;
  const float* xp = x + ((((size_t)b * TT + t) * 240) + y * 4) * 320 + xx * 4;
  float m = -3.4e38f;
  #pragma unroll
  for (int dy = 0; dy < 4; ++dy)
    #pragma unroll
    for (int dx = 0; dx < 4; ++dx)
      m = fmaxf(m, xp[dy * 320 + dx]);
  float v = F[OF_S1V + idx], ii = F[OF_S1I + idx];
  float vd = 0.9f * v + 0.1f * ii;              // v + dt*tau_mem*((0-v)+i)
  float z = (vd > 1.0f) ? 1.0f : 0.0f;
  F[OF_S1V + idx] = (1.0f - z) * vd;
  F[OF_S1I + idx] = 0.8f * ii + m;              // i - dt*tau_syn*i + inp
  H[OH_Z1 + idx] = (h16)z;
}

// conv1 (5x5, pad2, 1->16ch) via WMMA (A=weights 16x32, B=im2col 32x16) + LIF2
__global__ void k_conv1_lif2(h16* H, float* F, const float* __restrict__ c1b) {
  __shared__ h16 tile[8][128];                  // per-wave 5x20 halo of z1
  int lane = threadIdx.x & 31;
  int wib  = threadIdx.x >> 5;
  int wid  = blockIdx.x * 8 + wib;              // 4800 waves: b*300 + y*5 + xt
  int b = wid / 300, rem = wid % 300;
  int y = rem / 5, x0 = (rem % 5) << 4;
  const h16* z1 = H + OH_Z1 + (size_t)b * 4800;
  for (int i = lane; i < 100; i += 32) {
    int ry = i / 20, rx = i % 20;
    int gy = y + ry - 2, gx = x0 + rx - 2;
    h16 v = (h16)0.f;
    if (gy >= 0 && gy < 60 && gx >= 0 && gx < 80) v = z1[gy * 80 + gx];
    tile[wib][i] = v;
  }
  __syncthreads();
  // A fragment: one contiguous 32B per lane (pre-swizzled)
  v16h a = *(const v16h*)(H + OH_W1F + (size_t)lane * 16);
  int col = lane & 15;
  v16h bv;
  #pragma unroll
  for (int e = 0; e < 16; ++e) {
    int K = b_kmap(e, lane);
    h16 v = (h16)0.f;
    if (K < 25) { int dy = K / 5, dx = K % 5; v = tile[wib][dy * 20 + col + dx]; }
    bv[e] = v;
  }
  v8f c = {};
  c = __builtin_amdgcn_wmma_f32_16x16x32_f16(false, a, false, bv, (short)0, c, false, false);
  float* s2v = F + OF_S2V; float* s2i = F + OF_S2I;
  h16*   z2  = H + OH_Z2;
  #pragma unroll
  for (int v = 0; v < 8; ++v) {
    int co = v + ((lane >> 4) << 3);            // C layout: M = v + 8*(lane>=16)
    int xx = x0 + (lane & 15);                  // N = lane&15
    float pre = c[v] + c1b[co];
    size_t idx = (((size_t)b * 16 + co) * 60 + y) * 80 + xx;
    float vv = s2v[idx], ii = s2i[idx];
    float vd = 0.9f * vv + 0.1f * ii;
    float z = (vd > 1.0f) ? 1.0f : 0.0f;
    s2v[idx] = (1.0f - z) * vd;
    s2i[idx] = 0.8f * ii + pre;
    z2[idx]  = (h16)z;
  }
}

// 2x2 maxpool on z2 + LIF3 -> z3 (f16)
__global__ void k_pool2_lif3(h16* H, float* F) {
  int idx = blockIdx.x * blockDim.x + threadIdx.x;
  if (idx >= 307200) return;
  int b = idx / 19200, r = idx % 19200, c = r / 1200, r2 = r % 1200;
  int y = r2 / 40, xx = r2 % 40;
  const h16* z2 = H + OH_Z2 + (((size_t)b * 16 + c) * 60 + y * 2) * 80 + xx * 2;
  float m = fmaxf(fmaxf((float)z2[0], (float)z2[1]),
                  fmaxf((float)z2[80], (float)z2[81]));
  float v = F[OF_S3V + idx], ii = F[OF_S3I + idx];
  float vd = 0.9f * v + 0.1f * ii;
  float z = (vd > 1.0f) ? 1.0f : 0.0f;
  F[OF_S3V + idx] = (1.0f - z) * vd;
  F[OF_S3I + idx] = 0.8f * ii + m;
  H[OH_Z3 + idx] = (h16)z;
}

// conv2 (3x3, pad1, 16->32ch, x10) implicit GEMM via WMMA + LIF4.
// Weights staged to LDS by the Tensor Data Mover when available.
__global__ void k_conv2_lif4(h16* H, float* F, const float* __restrict__ c2b) {
  __shared__ h16 wlds[5120];                    // fragment-ordered conv2 weights (10KB)
#if __has_builtin(__builtin_amdgcn_tensor_load_to_lds)
  if (threadIdx.x < 32) {                       // one wave issues the TDM DMA
    unsigned long long ga = (unsigned long long)(H + OH_W2F);
    unsigned int lds_off = (unsigned int)(unsigned long long)(&wlds[0]);
    // D# group0: count=1, lds_addr, global_addr[56:0], type=2
    v4u g0 = { 1u, lds_off,
               (unsigned int)(ga & 0xFFFFFFFFu),
               (unsigned int)((ga >> 32) & 0x1FFFFFFu) | (2u << 30) };
    // D# group1: data_size=4B (code 2); 1-D tile: tensor_dim0=tile_dim0=2560 dwords
    const unsigned int ND = 2560u;              // 10240B / 4B
    v8i g1 = { (int)(2u << 16),                 // wgmask=0, data_size=2
               (int)((ND & 0xFFFFu) << 16),     // tensor_dim0[15:0] -> bits 63:48
               (int)((ND >> 16) | (1u << 16)),  // tensor_dim0[31:16], tensor_dim1=1
               (int)(ND << 16),                 // tile_dim0 -> bits 127:112
               (int)1,                          // tile_dim1=1, tile_dim2=0
               (int)ND,                         // tensor_dim0_stride[31:0]
               0, 0 };
    v4i g2 = { 0, 0, 0, 0 };
    v4i g3 = { 0, 0, 0, 0 };
#if __clang_major__ >= 23
    v8i g4 = { 0, 0, 0, 0, 0, 0, 0, 0 };
    __builtin_amdgcn_tensor_load_to_lds(g0, g1, g2, g3, g4, 0);
#else
    __builtin_amdgcn_tensor_load_to_lds(g0, g1, g2, g3, 0);
#endif
    __builtin_amdgcn_s_wait_tensorcnt((short)0);
  }
#else
  for (int i = threadIdx.x; i < 5120; i += 256) wlds[i] = H[OH_W2F + i];
#endif
  __syncthreads();
  int lane = threadIdx.x & 31;
  int wid  = blockIdx.x * 8 + (threadIdx.x >> 5); // 1200 waves: b*75 + mtile
  int b = wid / 75, p0 = (wid % 75) << 4;
  const h16* z3 = H + OH_Z3;
  int rowPos = p0 + (lane & 15);
  int ry = rowPos / 40, rx = rowPos % 40;
  v8f c0 = {}, c1 = {};
  #pragma unroll
  for (int ks = 0; ks < 5; ++ks) {
    v16h a;
    #pragma unroll
    for (int e = 0; e < 16; ++e) {
      int K = ks * 32 + a_kmap(e, lane);
      h16 v = (h16)0.f;
      if (K < 144) {
        int ci = K / 9, r = K % 9, kh = r / 3, kw = r % 3;
        int gy = ry + kh - 1, gx = rx + kw - 1;
        if (gy >= 0 && gy < 30 && gx >= 0 && gx < 40)
          v = z3[(((size_t)b * 16 + ci) * 30 + gy) * 40 + gx];
      }
      a[e] = v;
    }
    // B fragments: contiguous 32B per lane from LDS
    v16h b0 = *(const v16h*)&wlds[((ks * 2 + 0) * 32 + lane) * 16];
    v16h b1 = *(const v16h*)&wlds[((ks * 2 + 1) * 32 + lane) * 16];
    c0 = __builtin_amdgcn_wmma_f32_16x16x32_f16(false, a, false, b0, (short)0, c0, false, false);
    c1 = __builtin_amdgcn_wmma_f32_16x16x32_f16(false, a, false, b1, (short)0, c1, false, false);
  }
  float* s4v = F + OF_S4V; float* s4i = F + OF_S4I;
  h16*   z4  = H + OH_Z4;
  #pragma unroll
  for (int v = 0; v < 8; ++v) {
    int pm = p0 + v + ((lane >> 4) << 3);       // M = position
    int yy = pm / 40, xx = pm % 40;
    #pragma unroll
    for (int nt = 0; nt < 2; ++nt) {
      int co = nt * 16 + (lane & 15);           // N = channel
      float acc = nt ? c1[v] : c0[v];
      float pre = 10.0f * (acc + c2b[co]);
      size_t idx = (((size_t)b * 32 + co) * 30 + yy) * 40 + xx;
      float vv = s4v[idx], ii = s4i[idx];
      float vd = 0.9f * vv + 0.1f * ii;
      float z = (vd > 1.0f) ? 1.0f : 0.0f;
      s4v[idx] = (1.0f - z) * vd;
      s4i[idx] = 0.8f * ii + pre;
      z4[idx]  = (h16)z;
    }
  }
}

// 2x2 maxpool on z4 + LIF5 -> flattened spikes z5 [B,9600] (f16)
__global__ void k_pool2_lif5(h16* H, float* F) {
  int idx = blockIdx.x * blockDim.x + threadIdx.x;
  if (idx >= 153600) return;
  int b = idx / 9600, r = idx % 9600, c = r / 300, r2 = r % 300;
  int y = r2 / 20, xx = r2 % 20;
  const h16* z4 = H + OH_Z4 + (((size_t)b * 32 + c) * 30 + y * 2) * 40 + xx * 2;
  float m = fmaxf(fmaxf((float)z4[0], (float)z4[1]),
                  fmaxf((float)z4[40], (float)z4[41]));
  float v = F[OF_S5V + idx], ii = F[OF_S5I + idx];
  float vd = 0.9f * v + 0.1f * ii;
  float z = (vd > 1.0f) ? 1.0f : 0.0f;
  F[OF_S5V + idx] = (1.0f - z) * vd;
  F[OF_S5I + idx] = 0.8f * ii + m;
  H[OH_Z5 + idx] = (h16)z;                      // idx == b*9600 + (c*300+y*20+x)
}

// fc1: [16,9600]x[9600,512] WMMA GEMM, +bias, relu -> fc1a f32
__global__ void k_fc1(h16* H, float* F, const float* __restrict__ fc1b) {
  int lane = threadIdx.x & 31;
  int nt   = blockIdx.x * 8 + (threadIdx.x >> 5); // 32 N-tiles, one per wave
  const h16* z5 = H + OH_Z5;
  const h16* fw = H + OH_FWF + (size_t)nt * 300 * 512; // per-wave fragment base
  int rowB = lane & 15;                         // M = batch
  int colN = nt * 16 + (lane & 15);             // N = hidden unit
  const h16* za_base = z5 + (size_t)rowB * 9600 + ((lane >> 4) << 3);
  v8f c = {};
  for (int ks = 0; ks < 300; ++ks) {
    if (ks + 1 < 300)
      __builtin_prefetch(&fw[(size_t)(ks + 1) * 512 + lane * 16], 0, 1);
    // A fragment: two aligned 16B loads of spike row + shuffle into WMMA order
    const h16* za = za_base + ks * 32;
    v8h lo = *(const v8h*)za;
    v8h hi = *(const v8h*)(za + 16);
    v16h a = __builtin_shufflevector(lo, hi, 0, 1, 2, 3, 4, 5, 6, 7,
                                             8, 9, 10, 11, 12, 13, 14, 15);
    // B fragment: one contiguous 32B load per lane (wave reads 1KB contiguous)
    v16h bv = *(const v16h*)&fw[(size_t)ks * 512 + lane * 16];
    c = __builtin_amdgcn_wmma_f32_16x16x32_f16(false, a, false, bv, (short)0, c, false, false);
  }
  float* fc1a = F + OF_FC1A;
  #pragma unroll
  for (int v = 0; v < 8; ++v) {
    int m = v + ((lane >> 4) << 3);             // batch row
    float val = c[v] + (colN < 500 ? fc1b[colN] : 0.0f);
    fc1a[m * 512 + colN] = val > 0.0f ? val : 0.0f;
  }
}

// output LI cell: v update, i += relu(fc1) @ out_w.T, v_out = v + |min(v)|
__global__ void k_out(float* F, const float* __restrict__ outw, float* out, int t) {
  __shared__ float vsh[80];
  __shared__ float mn;
  int j = threadIdx.x;
  float vnew = 0.0f;
  if (j < 80) {
    int b = j / 5, o = j % 5;
    const float* xr = F + OF_FC1A + b * 512;
    const float* w  = outw + o * 500;
    float dot = 0.0f;
    for (int k = 0; k < 500; ++k) dot += xr[k] * w[k];
    float v = F[OF_SOV + j], ii = F[OF_SOI + j];
    vnew = 0.9f * v + 0.1f * ii;
    F[OF_SOV + j] = vnew;                       // state keeps unadjusted v
    F[OF_SOI + j] = 0.8f * ii + dot;
    vsh[j] = vnew;
  }
  __syncthreads();
  if (j == 0) {
    float m = vsh[0];
    for (int k = 1; k < 80; ++k) m = fminf(m, vsh[k]);
    mn = m;
  }
  __syncthreads();
  if (j < 80) out[(size_t)t * 80 + j] = vnew + fabsf(mn);
}

// ---------------- host launch ----------------
extern "C" void kernel_launch(void* const* d_in, const int* in_sizes, int n_in,
                              void* d_out, int out_size, void* d_ws, size_t ws_size,
                              hipStream_t stream) {
  const float* x    = (const float*)d_in[0];
  const float* c1w  = (const float*)d_in[1];
  const float* c1b  = (const float*)d_in[2];
  const float* c2w  = (const float*)d_in[3];
  const float* c2b  = (const float*)d_in[4];
  const float* fc1w = (const float*)d_in[5];
  const float* fc1b = (const float*)d_in[6];
  const float* outw = (const float*)d_in[7];
  float* out = (float*)d_out;

  float* F = (float*)d_ws;
  h16*   H = (h16*)((char*)d_ws + NF_TOTAL * sizeof(float));

  k_zero<<<2048, 256, 0, stream>>>(F);
  k_prep_w<<<(5632 + 255) / 256, 256, 0, stream>>>(H, c1w, c2w);
  k_prep_fwt<<<(int)(((size_t)32 * 300 * 32 * 16 + 255) / 256), 256, 0, stream>>>(H, fc1w);

  for (int t = 0; t < TT; ++t) {
    k_pool4_lif1<<<300,  256, 0, stream>>>(x, H, F, t);
    k_conv1_lif2<<<600,  256, 0, stream>>>(H, F, c1b);
    k_pool2_lif3<<<1200, 256, 0, stream>>>(H, F);
    k_conv2_lif4<<<150,  256, 0, stream>>>(H, F, c2b);
    k_pool2_lif5<<<600,  256, 0, stream>>>(H, F);
    k_fc1<<<4,           256, 0, stream>>>(H, F, fc1b);
    k_out<<<1,           128, 0, stream>>>(F, outw, out, t);
  }
}
